// Transformer_71914932404448
// MI455X (gfx1250) — matrix operand order repaired and checked by reading, not compiled, not run on hardware
//
#include <hip/hip_runtime.h>
#include <hip/hip_bf16.h>

#define EMB    768
#define HEADS  12
#define DKDIM  64
#define DFF    3072
#define BATCH  16
#define SEQ    577
#define TOK    (BATCH*SEQ)
#define LN_EPS 1e-5f

typedef __bf16 bf16;
typedef __attribute__((ext_vector_type(16))) __bf16 v16bf;
typedef __attribute__((ext_vector_type(8)))  __bf16 v8bf;
typedef __attribute__((ext_vector_type(4)))  __bf16 v4bf;
typedef __attribute__((ext_vector_type(8)))  float  v8f;
typedef __attribute__((ext_vector_type(4)))  float  v4f;
typedef __attribute__((ext_vector_type(4)))  unsigned int v4u;
typedef __attribute__((ext_vector_type(8)))  int    v8i;
typedef __attribute__((ext_vector_type(4)))  int    v4i;

union frag16 { v16bf v; v8bf h[2]; };

__device__ __forceinline__ v8f wmma_bf16(v16bf a, v16bf b, v8f c) {
  return __builtin_amdgcn_wmma_f32_16x16x32_bf16(false, a, false, b,
                                                 (short)0, c, false, false);
}

// Tensor Data Mover: DMA a 2D bf16 tile (tile_d1 rows x tile_d0 elems) from
// global (row stride = stride_elems) into LDS at lds_off, rows packed
// contiguously. Rows/cols beyond tensor_d1/tensor_d0 are zero-filled by HW.
// D# layout per CDNA5 ISA (2D: groups 2/3 zero).
// This toolchain exposes the 6-arg builtin: (g0, g1, g2, g3, g3b, cpol).
__device__ __forceinline__ void tdm_load_2d_bf16(
    unsigned lds_off, const void* gptr,
    unsigned tile_d0, unsigned tile_d1,
    unsigned tensor_d0, unsigned tensor_d1,
    unsigned stride_elems)
{
  unsigned long long ga = (unsigned long long)(uintptr_t)gptr;
  v4u g0;
  g0[0] = 1u;                                           // count=1 (valid), user mode
  g0[1] = lds_off;                                      // lds_addr[31:0]
  g0[2] = (unsigned)(ga & 0xFFFFFFFFu);                 // global_addr[31:0]
  g0[3] = (unsigned)((ga >> 32) & 0x1FFFFFFu) | (2u << 30); // addr[56:32], type=2
  v8i g1;
  g1[0] = (int)(1u << 16);                              // data_size=1 -> 2 bytes
  g1[1] = (int)((tensor_d0 & 0xFFFFu) << 16);           // tensor_dim0[15:0]
  g1[2] = (int)((tensor_d0 >> 16) | ((tensor_d1 & 0xFFFFu) << 16));
  g1[3] = (int)((tensor_d1 >> 16) | (tile_d0 << 16));   // tile_dim0
  g1[4] = (int)(tile_d1 & 0xFFFFu);                     // tile_dim1; tile_dim2=0
  g1[5] = (int)stride_elems;                            // tensor_dim0_stride[31:0]
  g1[6] = 0;
  g1[7] = 0;
  v4i z4; z4[0] = z4[1] = z4[2] = z4[3] = 0;
  v8i z8; for (int i = 0; i < 8; i++) z8[i] = 0;
  __builtin_amdgcn_tensor_load_to_lds(g0, g1, z4, z4, z8, 0);
}

// ---------------- LayerNorm: fp32 in -> bf16 out ----------------
__global__ __launch_bounds__(256)
void ln_kernel(const float* __restrict__ x, const float* __restrict__ g,
               const float* __restrict__ bta, bf16* __restrict__ out) {
  const int t = blockIdx.x;
  const float* xr = x + (size_t)t * EMB;
  const int i0 = threadIdx.x, i1 = threadIdx.x + 256, i2 = threadIdx.x + 512;
  float v0 = xr[i0], v1 = xr[i1], v2 = xr[i2];
  float s = v0 + v1 + v2;
  for (int m = 1; m < 32; m <<= 1) s += __shfl_xor(s, m, 32);
  __shared__ float red[8];
  __shared__ float red2[8];
  const int wave = threadIdx.x >> 5, lane = threadIdx.x & 31;
  if (lane == 0) red[wave] = s;
  __syncthreads();
  if (wave == 0) {
    float p = (lane < 8) ? red[lane] : 0.f;
    for (int m = 1; m < 8; m <<= 1) p += __shfl_xor(p, m, 32);
    if (lane == 0) red[0] = p;
  }
  __syncthreads();
  const float mu = red[0] * (1.0f / EMB);
  float d0 = v0 - mu, d1 = v1 - mu, d2 = v2 - mu;
  float q = d0*d0 + d1*d1 + d2*d2;
  for (int m = 1; m < 32; m <<= 1) q += __shfl_xor(q, m, 32);
  if (lane == 0) red2[wave] = q;
  __syncthreads();
  if (wave == 0) {
    float p = (lane < 8) ? red2[lane] : 0.f;
    for (int m = 1; m < 8; m <<= 1) p += __shfl_xor(p, m, 32);
    if (lane == 0) red2[0] = p;
  }
  __syncthreads();
  const float rstd = rsqrtf(red2[0] * (1.0f / EMB) + LN_EPS);
  bf16* orow = out + (size_t)t * EMB;
  orow[i0] = (bf16)(d0 * rstd * g[i0] + bta[i0]);
  orow[i1] = (bf16)(d1 * rstd * g[i1] + bta[i1]);
  orow[i2] = (bf16)(d2 * rstd * g[i2] + bta[i2]);
}

// ---------------- Tiled WMMA GEMM  C[T,N] = A[T,K] @ W[K,N] ----------------
// Block tile 128x128, K-step 32. 8 waves (2x4); each wave 64x32 = 4x2 WMMA tiles.
// bf16 A tiles are staged by the Tensor Data Mover (async DMA, TENSORcnt).
// MODE 0: +bias, scatter bf16 to [B,H,S,DK]   (QKV)
// MODE 1: +bias +resid(fp32,[T,N]), fp32 out  (O-proj / FFN2)
// MODE 2: +bias, exact GELU, bf16 out [T,N]   (FFN1)
template<typename AT, int MODE>
__global__ __launch_bounds__(256)
void gemm_kernel(const AT* __restrict__ A, const float* __restrict__ W,
                 const float* __restrict__ bias, const float* __restrict__ resid,
                 void* __restrict__ outp, int T, int K, int N) {
  __shared__ bf16 As[128 * 32];     // A tile, row-major [m][k]
  __shared__ bf16 Bt[128 * 32];     // W tile transposed: [n][k]
  const int tid  = threadIdx.x;
  const int wave = tid >> 5, lane = tid & 31;
  const int half = lane >> 4, ln16 = lane & 15;
  const int wm = wave >> 2, wn = wave & 3;
  const int t0 = blockIdx.y * 128;
  const int bn = blockIdx.x * 128;

  v8f acc[4][2];
  for (int i = 0; i < 4; i++)
    for (int j = 0; j < 2; j++)
      for (int r = 0; r < 8; r++) acc[i][j][r] = 0.f;

  for (int k0 = 0; k0 < K; k0 += 32) {
    // ---- stage A ----
    if constexpr (sizeof(AT) == 2) {
      // bf16: raw copy -> Tensor Data Mover DMA; rows beyond T zero-filled by HW
      if (wave == 0)
        tdm_load_2d_bf16((unsigned)(uintptr_t)As,
                         (const bf16*)A + (size_t)t0 * K + k0,
                         32u, 128u,
                         (unsigned)(K - k0), (unsigned)(T - t0),
                         (unsigned)K);
    } else {
      // fp32: load + convert manually
      int c0 = (tid & 7) * 4;
      int r  = tid >> 3;                       // 0..31
      for (int it = 0; it < 4; it++, r += 32) {
        int t = t0 + r;
        v4f av;
        if (t < T) av = *(const v4f*)((const float*)A + (size_t)t * K + k0 + c0);
        else { av[0] = av[1] = av[2] = av[3] = 0.f; }
        v4bf bv;
        bv[0] = (bf16)av[0]; bv[1] = (bf16)av[1];
        bv[2] = (bf16)av[2]; bv[3] = (bf16)av[3];
        *(v4bf*)(&As[r * 32 + c0]) = bv;
      }
    }
    // ---- stage W transposed (cvt to bf16), overlaps with TDM ----
    {
      int n0 = (tid & 31) * 4;
      int kr = tid >> 5;                       // 0..7
      for (int it = 0; it < 4; it++, kr += 8) {
        v4f wv = *(const v4f*)(W + (size_t)(k0 + kr) * N + bn + n0);
        Bt[(n0 + 0) * 32 + kr] = (bf16)wv[0];
        Bt[(n0 + 1) * 32 + kr] = (bf16)wv[1];
        Bt[(n0 + 2) * 32 + kr] = (bf16)wv[2];
        Bt[(n0 + 3) * 32 + kr] = (bf16)wv[3];
        if (k0 + 32 < K)
          __builtin_prefetch(W + (size_t)(k0 + 32 + kr) * N + bn + n0, 0, 0);
      }
    }
    if constexpr (sizeof(AT) == 2) {
      if (wave == 0) __builtin_amdgcn_s_wait_tensorcnt(0);
    }
    __syncthreads();
    // ---- fragments + WMMA ----
    frag16 bfr[2];
    for (int nt = 0; nt < 2; nt++) {
      int n = wn * 32 + nt * 16 + ln16;
      bfr[nt].h[0] = *(const v8bf*)(&Bt[n * 32 + half * 8]);
      bfr[nt].h[1] = *(const v8bf*)(&Bt[n * 32 + 16 + half * 8]);
    }
    for (int mt = 0; mt < 4; mt++) {
      int m = wm * 64 + mt * 16 + ln16;
      frag16 afr;
      afr.h[0] = *(const v8bf*)(&As[m * 32 + half * 8]);
      afr.h[1] = *(const v8bf*)(&As[m * 32 + 16 + half * 8]);
      for (int nt = 0; nt < 2; nt++)
        acc[mt][nt] = wmma_bf16(afr.v, bfr[nt].v, acc[mt][nt]);
    }
    __syncthreads();
  }
  // ---- epilogue ----
  for (int mt = 0; mt < 4; mt++)
    for (int nt = 0; nt < 2; nt++) {
      int col = bn + wn * 32 + nt * 16 + ln16;
      float bcol = bias[col];
      for (int r = 0; r < 8; r++) {
        int t = t0 + wm * 64 + mt * 16 + half * 8 + r;
        if (t >= T) continue;
        float v = acc[mt][nt][r] + bcol;
        if constexpr (MODE == 0) {
          int b = t / SEQ, s = t - b * SEQ;
          int hh = col >> 6, d = col & 63;
          ((bf16*)outp)[(((size_t)(b * HEADS + hh)) * SEQ + s) * DKDIM + d] = (bf16)v;
        } else if constexpr (MODE == 1) {
          v += resid[(size_t)t * N + col];
          ((float*)outp)[(size_t)t * N + col] = v;
        } else {
          float ge = 0.5f * v * (1.0f + erff(v * 0.70710678118f));
          ((bf16*)outp)[(size_t)t * N + col] = (bf16)ge;
        }
      }
    }
}

// ---------------- Flash attention (no 1/sqrt(dk) scaling, per reference) ----
// grid: (ceil(S/128), H, B); 8 waves, each owns 16 query rows.
// K tile staged by TDM; V staged manually (transposed for B-fragments).
__global__ __launch_bounds__(256)
void attn_kernel(const bf16* __restrict__ q, const bf16* __restrict__ k,
                 const bf16* __restrict__ v, float* __restrict__ ctx) {
  __shared__ bf16 Ks[32 * 64];        // [key][d]
  __shared__ bf16 Vt[64 * 32];        // transposed: [d][key]
  __shared__ bf16 Pb[8][16 * 32];     // per-wave P tile [row][key]
  const int tid  = threadIdx.x;
  const int wave = tid >> 5, lane = tid & 31;
  const int half = lane >> 4, ln16 = lane & 15;
  const int hh = blockIdx.y, b = blockIdx.z;
  const int q0 = blockIdx.x * 128 + wave * 16;
  const size_t bh = ((size_t)b * HEADS + hh) * SEQ;

  frag16 qf[2];
  {
    int qr = q0 + ln16; if (qr >= SEQ) qr = SEQ - 1;
    const bf16* qrow = q + (bh + qr) * DKDIM;
    for (int ks = 0; ks < 2; ks++) {
      qf[ks].h[0] = *(const v8bf*)(qrow + ks * 32 + half * 8);
      qf[ks].h[1] = *(const v8bf*)(qrow + ks * 32 + 16 + half * 8);
    }
  }
  float mstate[8], lstate[8];
  v8f acc[4];
  for (int r = 0; r < 8; r++) { mstate[r] = -1e30f; lstate[r] = 0.f; }
  for (int t = 0; t < 4; t++)
    for (int r = 0; r < 8; r++) acc[t][r] = 0.f;

  const int nkb = (SEQ + 31) / 32;
  for (int kb = 0; kb < nkb; kb++) {
    // ---- stage K tile via TDM (rows >= SEQ zero-filled; masked below anyway)
    if (wave == 0)
      tdm_load_2d_bf16((unsigned)(uintptr_t)Ks,
                       k + (bh + (size_t)kb * 32) * DKDIM,
                       64u, 32u,
                       64u, (unsigned)(SEQ - kb * 32),
                       64u);
    // ---- stage transposed V tile manually (overlaps with TDM) ----
    {
      int row = tid >> 3;              // 0..31
      int d0  = (tid & 7) * 8;
      int key = kb * 32 + row;
      v8bf vv;
      if (key < SEQ) vv = *(const v8bf*)(v + (bh + key) * DKDIM + d0);
      else for (int e = 0; e < 8; e++) vv[e] = (bf16)0.f;
      for (int e = 0; e < 8; e++) Vt[(d0 + e) * 32 + row] = vv[e];
    }
    if (wave == 0) __builtin_amdgcn_s_wait_tensorcnt(0);
    __syncthreads();
    // scores = Q . K^T (two 16-key n-tiles, two k-steps over dk)
    v8f sc[2];
    for (int nt = 0; nt < 2; nt++) {
      for (int r = 0; r < 8; r++) sc[nt][r] = 0.f;
      for (int ks = 0; ks < 2; ks++) {
        frag16 kf;
        const int base = (nt * 16 + ln16) * 64 + ks * 32 + half * 8;
        kf.h[0] = *(const v8bf*)(&Ks[base]);
        kf.h[1] = *(const v8bf*)(&Ks[base + 16]);
        sc[nt] = wmma_bf16(qf[ks].v, kf.v, sc[nt]);
      }
      int key = kb * 32 + nt * 16 + ln16;
      if (key >= SEQ)
        for (int r = 0; r < 8; r++) sc[nt][r] = -1e30f;
    }
    // online softmax (rows live in 16-lane halves)
    float scale[8];
    for (int r = 0; r < 8; r++) {
      float mx = fmaxf(sc[0][r], sc[1][r]);
      for (int msk = 1; msk < 16; msk <<= 1) mx = fmaxf(mx, __shfl_xor(mx, msk, 32));
      float mnew = fmaxf(mstate[r], mx);
      scale[r] = __expf(mstate[r] - mnew);
      float p0 = __expf(sc[0][r] - mnew);
      float p1 = __expf(sc[1][r] - mnew);
      sc[0][r] = p0; sc[1][r] = p1;
      float rs = p0 + p1;
      for (int msk = 1; msk < 16; msk <<= 1) rs += __shfl_xor(rs, msk, 32);
      lstate[r] = lstate[r] * scale[r] + rs;
      mstate[r] = mnew;
    }
    for (int t = 0; t < 4; t++)
      for (int r = 0; r < 8; r++) acc[t][r] *= scale[r];
    // P -> LDS (bf16), reload as A-fragment
    for (int r = 0; r < 8; r++) {
      Pb[wave][(half * 8 + r) * 32 + ln16]      = (bf16)sc[0][r];
      Pb[wave][(half * 8 + r) * 32 + 16 + ln16] = (bf16)sc[1][r];
    }
    frag16 pf;
    pf.h[0] = *(const v8bf*)(&Pb[wave][ln16 * 32 + half * 8]);
    pf.h[1] = *(const v8bf*)(&Pb[wave][ln16 * 32 + 16 + half * 8]);
    // ctx += P @ V
    for (int t = 0; t < 4; t++) {
      frag16 vf;
      const int base = (t * 16 + ln16) * 32 + half * 8;
      vf.h[0] = *(const v8bf*)(&Vt[base]);
      vf.h[1] = *(const v8bf*)(&Vt[base + 16]);
      acc[t] = wmma_bf16(pf.v, vf.v, acc[t]);
    }
    __syncthreads();
  }
  // write ctx fp32 [B,S,EMB]
  for (int t = 0; t < 4; t++)
    for (int r = 0; r < 8; r++) {
      int qr = q0 + half * 8 + r;
      if (qr < SEQ) {
        float inv = 1.0f / lstate[r];
        ctx[((size_t)b * SEQ + qr) * EMB + hh * DKDIM + t * 16 + ln16] = acc[t][r] * inv;
      }
    }
}

extern "C" void kernel_launch(void* const* d_in, const int* in_sizes, int n_in,
                              void* d_out, int out_size, void* d_ws, size_t ws_size,
                              hipStream_t stream) {
  (void)in_sizes; (void)n_in; (void)out_size; (void)ws_size;
  const float* x   = (const float*)d_in[0];
  const float* wq  = (const float*)d_in[1];
  const float* bq  = (const float*)d_in[2];
  const float* wk  = (const float*)d_in[3];
  const float* bk  = (const float*)d_in[4];
  const float* wv  = (const float*)d_in[5];
  const float* bv  = (const float*)d_in[6];
  const float* wo  = (const float*)d_in[7];
  const float* bo  = (const float*)d_in[8];
  const float* w1  = (const float*)d_in[9];
  const float* bf1 = (const float*)d_in[10];
  const float* w2  = (const float*)d_in[11];
  const float* bf2 = (const float*)d_in[12];
  const float* g1  = (const float*)d_in[13];
  const float* b1  = (const float*)d_in[14];
  const float* g2  = (const float*)d_in[15];
  const float* b2  = (const float*)d_in[16];

  char* ws = (char*)d_ws;
  const size_t SZ_BF = (size_t)TOK * EMB * sizeof(bf16);   // 14.2 MB
  const size_t SZ_F  = (size_t)TOK * EMB * sizeof(float);  // 28.4 MB
  bf16*  hbuf = (bf16*)(ws);                        // post-LN1 (bf16)
  bf16*  qb   = (bf16*)(ws + SZ_BF);                // q [B,H,S,DK]
  bf16*  kbuf = (bf16*)(ws + 2 * SZ_BF);            // k
  bf16*  vb   = (bf16*)(ws + 3 * SZ_BF);            // v
  float* ctx  = (float*)(ws + 4 * SZ_BF);           // attention output (fp32)
  float* x2   = (float*)(ws + 4 * SZ_BF + SZ_F);    // x + attn (fp32)
  bf16*  h2   = (bf16*)(ws + 4 * SZ_BF + 2 * SZ_F); // post-LN2 (bf16)
  bf16*  ffh  = (bf16*)(ws);                        // FFN hidden, reuses region0 (== TOK*DFF*2 bytes)

  dim3 blk(256);
  ln_kernel<<<TOK, blk, 0, stream>>>(x, g1, b1, hbuf);

  dim3 gq((EMB + 127) / 128, (TOK + 127) / 128);
  gemm_kernel<bf16, 0><<<gq, blk, 0, stream>>>(hbuf, wq, bq, nullptr, qb,   TOK, EMB, EMB);
  gemm_kernel<bf16, 0><<<gq, blk, 0, stream>>>(hbuf, wk, bk, nullptr, kbuf, TOK, EMB, EMB);
  gemm_kernel<bf16, 0><<<gq, blk, 0, stream>>>(hbuf, wv, bv, nullptr, vb,   TOK, EMB, EMB);

  dim3 ga((SEQ + 127) / 128, HEADS, BATCH);
  attn_kernel<<<ga, blk, 0, stream>>>(qb, kbuf, vb, ctx);

  gemm_kernel<float, 1><<<gq, blk, 0, stream>>>(ctx, wo, bo, x, x2, TOK, EMB, EMB);

  ln_kernel<<<TOK, blk, 0, stream>>>(x2, g2, b2, h2);

  dim3 gf1((DFF + 127) / 128, (TOK + 127) / 128);
  gemm_kernel<bf16, 2><<<gf1, blk, 0, stream>>>(h2, w1, bf1, nullptr, ffh, TOK, EMB, DFF);
  gemm_kernel<bf16, 1><<<gq, blk, 0, stream>>>(ffh, w2, bf2, x2, d_out, TOK, DFF, EMB);
}
